// MultiHeadAttention_24618752540916
// MI455X (gfx1250) — compile-verified, matrix operand order checked by reading, not run on hardware
//
#include <hip/hip_runtime.h>

// ---------------------------------------------------------------------------
// MHA forward for MI455X (gfx1250): fp16 WMMA, fp32 accumulate, flash fusion.
// Fragment loads vectorized to b128 (fragment K-pairs are contiguous in the
// ISA 16x16x32 f16 layout); store addresses hoisted so unrolled offsets fold
// into instruction immediates.
// ---------------------------------------------------------------------------

typedef __attribute__((ext_vector_type(16))) _Float16 v16h;
typedef __attribute__((ext_vector_type(8)))  float    v8f;
typedef __attribute__((ext_vector_type(4)))  uint32_t v4u;
typedef __attribute__((ext_vector_type(2)))  uint32_t v2u;
typedef __attribute__((ext_vector_type(4)))  float    v4f;

#define DIM_   512
#define HEADS_ 8
#define DH_    64
#define B_     4
#define S_     2048
#define MROWS_ (B_ * S_)

union Frag16 { v16h h; uint32_t u[8]; v4u q[2]; };
union FragF  { v8f  v; float    f[8]; };
union U2H    { v2u  u2; _Float16 hf[4]; };

// ---------------------------------------------------------------------------
// Kernel 1: Y = x @ W^T + b, fp32 -> fp16, head-split [B,H,S,Dh].
// Block = 128 threads (4 waves); wave computes a 16x16 tile; K loop over 512.
// ---------------------------------------------------------------------------
__global__ __launch_bounds__(128) void qkv_proj_kernel(
    const float* __restrict__ x, const float* __restrict__ W,
    const float* __restrict__ bias, _Float16* __restrict__ dst)
{
    const int tid  = threadIdx.x;
    const int wave = tid >> 5;
    const int lane = tid & 31;
    const int l15  = lane & 15;
    const int klo  = (lane < 16) ? 0 : 8;   // fragment K base for this lane

    const int mt = blockIdx.x;               // 0..511
    const int nt = blockIdx.y * 4 + wave;    // 0..31

    const float* xrow = x + (size_t)(mt * 16 + l15) * DIM_ + klo;
    const float* wrow = W + (size_t)(nt * 16 + l15) * DIM_ + klo;

    FragF c; c.v = (v8f){0.f, 0.f, 0.f, 0.f, 0.f, 0.f, 0.f, 0.f};

#pragma unroll 4
    for (int kk = 0; kk < DIM_; kk += 32) {
        const v4f fa0 = *reinterpret_cast<const v4f*>(xrow + kk);
        const v4f fa1 = *reinterpret_cast<const v4f*>(xrow + kk + 4);
        const v4f fa2 = *reinterpret_cast<const v4f*>(xrow + kk + 16);
        const v4f fa3 = *reinterpret_cast<const v4f*>(xrow + kk + 20);
        const v4f fb0 = *reinterpret_cast<const v4f*>(wrow + kk);
        const v4f fb1 = *reinterpret_cast<const v4f*>(wrow + kk + 4);
        const v4f fb2 = *reinterpret_cast<const v4f*>(wrow + kk + 16);
        const v4f fb3 = *reinterpret_cast<const v4f*>(wrow + kk + 20);
        Frag16 a, b;
#pragma unroll
        for (int e = 0; e < 4; ++e) {
            a.h[e]      = (_Float16)fa0[e];
            a.h[4 + e]  = (_Float16)fa1[e];
            a.h[8 + e]  = (_Float16)fa2[e];
            a.h[12 + e] = (_Float16)fa3[e];
            b.h[e]      = (_Float16)fb0[e];
            b.h[4 + e]  = (_Float16)fb1[e];
            b.h[8 + e]  = (_Float16)fb2[e];
            b.h[12 + e] = (_Float16)fb3[e];
        }
        c.v = __builtin_amdgcn_wmma_f32_16x16x32_f16(
                  false, a.h, false, b.h, (short)0, c.v, false, false);
    }

    const int   ncol = nt * 16 + l15;
    const float bvv  = bias[ncol];
    const int   hh   = ncol >> 6;
    const int   d    = ncol & 63;

    // 16-row tile never straddles a batch boundary (S_ % 16 == 0).
    const int rowbase = mt * 16 + ((lane < 16) ? 0 : 8);
    const int bb      = rowbase >> 11;       // / S_
    const int ss      = rowbase & (S_ - 1);
    _Float16* dp = dst + (((size_t)bb * HEADS_ + hh) * S_ + ss) * DH_ + d;
#pragma unroll
    for (int r = 0; r < 8; ++r)
        dp[r * DH_] = (_Float16)(c.f[r] + bvv);   // constant offsets fold
}

// ---------------------------------------------------------------------------
// Kernel 2: flash attention. Block = 128 threads = 4 waves, one (b,h) per
// blockIdx.y, 64 query rows per block. K tile row-major / V tile transposed
// in LDS; all fragments read as 2x b128; online softmax in registers.
// ---------------------------------------------------------------------------
__global__ __launch_bounds__(128) void flash_attn_kernel(
    const _Float16* __restrict__ Qh, const _Float16* __restrict__ Kh,
    const _Float16* __restrict__ Vh, float* __restrict__ out)
{
    __shared__ __align__(16) _Float16 Kt[32][72];     // 144B rows (16B-mult)
    __shared__ __align__(16) _Float16 Vt[64][56];     // 112B rows, transposed
    __shared__ __align__(16) _Float16 Pt[4][16][40];  // 80B rows, per-wave

    const int tid  = threadIdx.x;
    const int wave = tid >> 5;
    const int lane = tid & 31;
    const int l15  = lane & 15;
    const int klo  = (lane < 16) ? 0 : 8;

    const int    bh     = blockIdx.y;          // b*HEADS + h
    const int    bb     = bh >> 3;
    const int    hh     = bh & 7;
    const int    q0     = blockIdx.x * 64;
    const size_t kvbase = (size_t)bh * S_ * DH_;
    const _Float16* __restrict__ ksrc = Kh + kvbase;
    const _Float16* __restrict__ vsrc = Vh + kvbase;

    // --- Q fragments once: 2 A-frags (d=0..31, 32..63), 4x global b128 ---
    Frag16 qf[2];
    {
        const v4u* qp = reinterpret_cast<const v4u*>(
            Qh + kvbase + (size_t)(q0 + wave * 16 + l15) * DH_);
        const int o = klo >> 3;                // 0 or 1 (v4u units)
#pragma unroll
        for (int t = 0; t < 2; ++t) {
            qf[t].q[0] = qp[4 * t + o];
            qf[t].q[1] = qp[4 * t + o + 2];
        }
    }

    FragF o4[4];
#pragma unroll
    for (int t = 0; t < 4; ++t)
        o4[t].v = (v8f){0.f, 0.f, 0.f, 0.f, 0.f, 0.f, 0.f, 0.f};
    float m_i[8], l_i[8];
#pragma unroll
    for (int r = 0; r < 8; ++r) { m_i[r] = -1e30f; l_i[r] = 0.f; }

    const float SC = 0.125f;   // 1/sqrt(64)

    for (int j0 = 0; j0 < S_; j0 += 32) {
        __syncthreads();
        // ---- cooperative staging: K row-major, V transposed ----
#pragma unroll
        for (int i = 0; i < 4; ++i) {
            const int g   = tid + i * 128;     // 512 groups of 4 halves
            const int row = g >> 4;
            const int c4  = (g & 15) * 4;
            const size_t src = (size_t)(j0 + row) * DH_ + c4;
            *reinterpret_cast<v2u*>(&Kt[row][c4]) =
                *reinterpret_cast<const v2u*>(ksrc + src);
            U2H v2; v2.u2 = *reinterpret_cast<const v2u*>(vsrc + src);
            Vt[c4 + 0][row] = v2.hf[0];
            Vt[c4 + 1][row] = v2.hf[1];
            Vt[c4 + 2][row] = v2.hf[2];
            Vt[c4 + 3][row] = v2.hf[3];
        }
        if (j0 + 32 < S_) {
            __builtin_prefetch(ksrc + (size_t)(j0 + 32) * DH_ + tid * 16, 0, 1);
            __builtin_prefetch(vsrc + (size_t)(j0 + 32) * DH_ + tid * 16, 0, 1);
        }
        __syncthreads();

        // ---- scores: S[16x32] = Q @ K^T (two C-frags, 4 WMMA) ----
        FragF s01[2];
#pragma unroll
        for (int jt = 0; jt < 2; ++jt) {
            const int krow = jt * 16 + l15;
            Frag16 kf0, kf1;
            kf0.q[0] = *reinterpret_cast<const v4u*>(&Kt[krow][klo]);
            kf0.q[1] = *reinterpret_cast<const v4u*>(&Kt[krow][klo + 16]);
            kf1.q[0] = *reinterpret_cast<const v4u*>(&Kt[krow][klo + 32]);
            kf1.q[1] = *reinterpret_cast<const v4u*>(&Kt[krow][klo + 48]);
            v8f acc = (v8f){0.f, 0.f, 0.f, 0.f, 0.f, 0.f, 0.f, 0.f};
            acc = __builtin_amdgcn_wmma_f32_16x16x32_f16(
                      false, qf[0].h, false, kf0.h, (short)0, acc, false, false);
            acc = __builtin_amdgcn_wmma_f32_16x16x32_f16(
                      false, qf[1].h, false, kf1.h, (short)0, acc, false, false);
            s01[jt].v = acc;
        }

        // ---- online softmax ----
#pragma unroll
        for (int r = 0; r < 8; ++r) {
            const float a  = s01[0].f[r] * SC;
            const float bv = s01[1].f[r] * SC;
            float mx = fmaxf(a, bv);
            mx = fmaxf(mx, __shfl_xor(mx, 1));
            mx = fmaxf(mx, __shfl_xor(mx, 2));
            mx = fmaxf(mx, __shfl_xor(mx, 4));
            mx = fmaxf(mx, __shfl_xor(mx, 8));
            const float nm   = fmaxf(m_i[r], mx);
            const float corr = __expf(m_i[r] - nm);
            m_i[r] = nm;
            const float p0 = __expf(a - nm);
            const float p1 = __expf(bv - nm);
            float rs = p0 + p1;
            rs += __shfl_xor(rs, 1);
            rs += __shfl_xor(rs, 2);
            rs += __shfl_xor(rs, 4);
            rs += __shfl_xor(rs, 8);
            l_i[r] = l_i[r] * corr + rs;
#pragma unroll
            for (int t = 0; t < 4; ++t) o4[t].f[r] *= corr;
            const int prow = r + ((lane < 16) ? 0 : 8);
            Pt[wave][prow][l15]      = (_Float16)p0;
            Pt[wave][prow][16 + l15] = (_Float16)p1;
        }

        // ---- P: C-layout -> A-layout via per-wave LDS (2x b128) ----
        Frag16 pf;
        pf.q[0] = *reinterpret_cast<const v4u*>(&Pt[wave][l15][klo]);
        pf.q[1] = *reinterpret_cast<const v4u*>(&Pt[wave][l15][klo + 16]);

        // ---- O += P @ V (4 d-tiles, 4 WMMA) ----
#pragma unroll
        for (int t = 0; t < 4; ++t) {
            const int d = t * 16 + l15;
            Frag16 vf;
            vf.q[0] = *reinterpret_cast<const v4u*>(&Vt[d][klo]);
            vf.q[1] = *reinterpret_cast<const v4u*>(&Vt[d][klo + 16]);
            o4[t].v = __builtin_amdgcn_wmma_f32_16x16x32_f16(
                          false, pf.h, false, vf.h, (short)0, o4[t].v, false, false);
        }
    }

    // ---- normalize + store (base hoisted; r/t offsets fold) ----
    float* op = out + ((size_t)bb * S_ + q0 + wave * 16 + ((lane < 16) ? 0 : 8)) * DIM_
                    + hh * DH_ + l15;
#pragma unroll
    for (int r = 0; r < 8; ++r) {
        const float inv = 1.0f / l_i[r];
#pragma unroll
        for (int t = 0; t < 4; ++t)
            op[r * DIM_ + t * 16] = o4[t].f[r] * inv;
    }
}

// ---------------------------------------------------------------------------
extern "C" void kernel_launch(void* const* d_in, const int* in_sizes, int n_in,
                              void* d_out, int out_size, void* d_ws, size_t ws_size,
                              hipStream_t stream)
{
    const float* q  = (const float*)d_in[0];
    const float* k  = (const float*)d_in[1];
    const float* v  = (const float*)d_in[2];
    const float* Wq = (const float*)d_in[3];
    const float* bq = (const float*)d_in[4];
    const float* Wk = (const float*)d_in[5];
    const float* bk = (const float*)d_in[6];
    const float* Wv = (const float*)d_in[7];
    const float* bv = (const float*)d_in[8];
    float* out = (float*)d_out;

    const size_t per = (size_t)B_ * HEADS_ * S_ * DH_;
    _Float16* Qh = (_Float16*)d_ws;
    _Float16* Kh = Qh + per;
    _Float16* Vh = Kh + per;

    dim3 pg(MROWS_ / 16, DIM_ / 64);
    dim3 pb(128);
    qkv_proj_kernel<<<pg, pb, 0, stream>>>(q, Wq, bq, Qh);
    qkv_proj_kernel<<<pg, pb, 0, stream>>>(k, Wk, bk, Kh);
    qkv_proj_kernel<<<pg, pb, 0, stream>>>(v, Wv, bv, Vh);

    dim3 fg(S_ / 64, B_ * HEADS_);
    dim3 fb(128);
    flash_attn_kernel<<<fg, fb, 0, stream>>>(Qh, Kh, Vh, out);
}
